// FiGNNBlock_54065048322681
// MI455X (gfx1250) — matrix operand order repaired and checked by reading, not compiled, not run on hardware
//
#include <hip/hip_runtime.h>
#include <hip/hip_bf16.h>

typedef __attribute__((ext_vector_type(16))) __bf16 v16bf;
typedef __attribute__((ext_vector_type(8)))  __bf16 v8bf;
typedef __attribute__((ext_vector_type(8)))  float  v8f;

#define NF 40
#define DD 64
#define TB 8                  // batch rows per tile
#define ROWS (TB*NF)          // 320 node rows per tile
#define BLOCK 640
#define NWAVES (BLOCK/32)     // 20 waves
#define NEG_SLOPE 0.01f

// padded LDS strides (bank-conflict avoidance + 16B alignment)
#define SH_ROW 68             // f32 elements per h row
#define G_ROW  72             // bf16 elements per g row (j padded 40->64, stride 72)
#define T2_ROW 72             // bf16 elements per aggr/a row
#define HO_J   72             // bf16 j-stride of transposed h_out
#define HO_SLAB (DD*HO_J + 8) // 4616 bf16 per batch-row slab

// ---- workspace layout (bf16 elements) ----
#define ASW_ELEMS (NF*4*2*32*16)   // 163840 per A-swizzled weight (W_out / W_in)
#define BSW_ELEMS (12*2*32*16)     // 12288 per B-swizzled weight (W_ih / W_hh)
#define ASW_OUT_OFF 0
#define ASW_IN_OFF  (ASW_ELEMS)
#define BSW_IH_OFF  (2*ASW_ELEMS)
#define BSW_HH_OFF  (2*ASW_ELEMS + BSW_ELEMS)
#define WS_TOTAL    (2*ASW_ELEMS + 2*BSW_ELEMS)

// ---- LDS layout (bytes, all 16B aligned) ----
#define SH_OFF  0                       // f32 h  [320][68]            = 87040
#define SG_OFF  87040                   // bf16 g [8*40][72]           = 46080
#define HOT_OFF 133120                  // bf16 h_out^T [8][4616]      = 73856
#define ST2_OFF 206976                  // bf16 aggr/a [8*40][72]      = 46080
#define SS_OFF  253056                  // f32 s[320]                  = 1280
#define STT_OFF 254336                  // f32 t[320]                  = 1280
#define LDS_BYTES 255616

// Pre-swizzle the four weight matrices into exact WMMA fragment order.
// A-layout (16-bit, 16x32): lane l row M=l%16; elem e -> K = 16*(l/16)+e
// B-layout (16-bit, 32x16): lane l col N=l%16; elem e -> K = 16*(l/16)+e
__global__ void fignn_prep_weights(const float* __restrict__ Wout,
                                   const float* __restrict__ Win,
                                   const float* __restrict__ Wih,
                                   const float* __restrict__ Whh,
                                   __bf16* __restrict__ ws) {
  int idx = blockIdx.x * 256 + threadIdx.x;
  if (idx < 2 * ASW_ELEMS) {
    const float* W = (idx < ASW_ELEMS) ? Wout : Win;
    __bf16* dst = ws + ((idx < ASW_ELEMS) ? ASW_OUT_OFF : ASW_IN_OFF);
    int i = idx % ASW_ELEMS;
    int e = i & 15, lane = (i >> 4) & 31, ks = (i >> 9) & 1, mt = (i >> 10) & 3, f = i >> 12;
    int d = 16 * mt + (lane & 15);
    int k = 32 * ks + 16 * (lane >> 4) + e;
    dst[i] = (__bf16)W[(f * DD + d) * DD + k];        // W[f,d,e]: A rows = d, K = e
  } else if (idx < 2 * ASW_ELEMS + 2 * BSW_ELEMS) {
    int j = idx - 2 * ASW_ELEMS;
    const float* W = (j < BSW_ELEMS) ? Wih : Whh;
    __bf16* dst = ws + ((j < BSW_ELEMS) ? BSW_IH_OFF : BSW_HH_OFF);
    int i = j % BSW_ELEMS;
    int e = i & 15, lane = (i >> 4) & 31, ks = (i >> 9) & 1, nc = i >> 10;
    int nrow = 16 * nc + (lane & 15);
    int k = 32 * ks + 16 * (lane >> 4) + e;
    dst[i] = (__bf16)W[nrow * DD + k];                // W[n,k]: B cols = n, K = k
  }
}

__device__ __forceinline__ v8f wmma_bf16(v16bf a, v16bf b, v8f c) {
  return __builtin_amdgcn_wmma_f32_16x16x32_bf16(false, a, false, b, (short)0, c,
                                                 false, false);
}

// 16 contiguous bf16 (two aligned 16B LDS loads)
__device__ __forceinline__ v16bf frag16_bf16(const __bf16* p) {
  v8bf lo = *(const v8bf*)p;
  v8bf hi = *(const v8bf*)(p + 8);
  return __builtin_shufflevector(lo, hi, 0, 1, 2, 3, 4, 5, 6, 7,
                                 8, 9, 10, 11, 12, 13, 14, 15);
}

// A-fragment from bf16 row: run [k0,k0+16), k0 = 32ks+16khalf
__device__ __forceinline__ v16bf a_frag_bf16(const __bf16* rowp, int ks, int khalf) {
  int k0 = 32 * ks + 16 * khalf;
  v8bf lo = *(const v8bf*)(rowp + k0);
  v8bf hi = *(const v8bf*)(rowp + k0 + 8);
  return __builtin_shufflevector(lo, hi, 0, 1, 2, 3, 4, 5, 6, 7,
                                 8, 9, 10, 11, 12, 13, 14, 15);
}

// A-fragment from f32 row (convert to bf16): same K pattern
__device__ __forceinline__ v16bf a_frag_f32(const float* rowp, int ks, int khalf) {
  int k0 = 32 * ks + 16 * khalf;
  const float4* p = (const float4*)(rowp + k0);
  const float4* q = (const float4*)(rowp + k0 + 8);
  float4 x0 = p[0], x1 = p[1], x2 = q[0], x3 = q[1];
  v16bf r;
  r[0] = (__bf16)x0.x;  r[1] = (__bf16)x0.y;  r[2] = (__bf16)x0.z;  r[3] = (__bf16)x0.w;
  r[4] = (__bf16)x1.x;  r[5] = (__bf16)x1.y;  r[6] = (__bf16)x1.z;  r[7] = (__bf16)x1.w;
  r[8] = (__bf16)x2.x;  r[9] = (__bf16)x2.y;  r[10] = (__bf16)x2.z; r[11] = (__bf16)x2.w;
  r[12] = (__bf16)x3.x; r[13] = (__bf16)x3.y; r[14] = (__bf16)x3.z; r[15] = (__bf16)x3.w;
  return r;
}

// B-fragment (16 contiguous K values) from f32, converted
__device__ __forceinline__ v16bf b_frag_f32(const float* p) {
  const float4* q = (const float4*)p;
  float4 x0 = q[0], x1 = q[1], x2 = q[2], x3 = q[3];
  v16bf r;
  r[0] = (__bf16)x0.x;  r[1] = (__bf16)x0.y;  r[2] = (__bf16)x0.z;  r[3] = (__bf16)x0.w;
  r[4] = (__bf16)x1.x;  r[5] = (__bf16)x1.y;  r[6] = (__bf16)x1.z;  r[7] = (__bf16)x1.w;
  r[8] = (__bf16)x2.x;  r[9] = (__bf16)x2.y;  r[10] = (__bf16)x2.z; r[11] = (__bf16)x2.w;
  r[12] = (__bf16)x3.x; r[13] = (__bf16)x3.y; r[14] = (__bf16)x3.z; r[15] = (__bf16)x3.w;
  return r;
}

__global__ __launch_bounds__(BLOCK)
void fignn_fused(const float* __restrict__ feat,
                 const float* __restrict__ w_attn,
                 const float* __restrict__ bias_p,
                 const float* __restrict__ b_ih,
                 const float* __restrict__ b_hh,
                 const __bf16* __restrict__ ws,
                 float* __restrict__ out) {
  extern __shared__ char smem[];
  float*  sH  = (float*)(smem + SH_OFF);    // h, padded rows (in-place update)
  __bf16* sG  = (__bf16*)(smem + SG_OFF);   // softmax graph, K-padded to 64
  __bf16* hoT = (__bf16*)(smem + HOT_OFF);  // h_out transposed [bi][d][j], j-padded
  __bf16* sT2 = (__bf16*)(smem + ST2_OFF);  // aggr, then a (same buffer)
  float*  sS  = (float*)(smem + SS_OFF);
  float*  sTt = (float*)(smem + STT_OFF);

  const int tid   = threadIdx.x;
  const int lane  = tid & 31;
  const int wv    = tid >> 5;
  const int khalf = lane >> 4;
  const int lmod  = lane & 15;
  const size_t tilebase = (size_t)blockIdx.x * TB * NF * DD;

  const __bf16* AswOut = ws + ASW_OUT_OFF;
  const __bf16* AswIn  = ws + ASW_IN_OFF;
  const __bf16* BswIh  = ws + BSW_IH_OFF;
  const __bf16* BswHh  = ws + BSW_HH_OFF;

  const bool biok = (lmod < TB);
  const int  bic  = biok ? lmod : 0;       // clamped batch lane for B-fragments
  const v16bf vzero = {};

  // ---- zero hoT (padding regions must read as 0 in aggr GEMM) ----
  for (int i = tid; i < TB * HO_SLAB / 2; i += BLOCK) ((uint*)hoT)[i] = 0u;

  // ---- load feat tile into padded sH rows: h0 = feat ----
  {
    const float4* f4 = (const float4*)(feat + tilebase);
    for (int i = tid; i < ROWS * (DD / 4); i += BLOCK) {
      int r = i >> 4, c = i & 15;
      ((float4*)(sH + r * SH_ROW))[c] = f4[i];
    }
  }
  __syncthreads();

  // ---- attention: s,t dot products ----
  for (int r = tid; r < ROWS; r += BLOCK) {
    const float* hp = sH + r * SH_ROW;
    float s = 0.f, t = 0.f;
    #pragma unroll 8
    for (int e = 0; e < DD; ++e) {
      float v = hp[e];
      s = fmaf(v, w_attn[e], s);
      t = fmaf(v, w_attn[DD + e], t);
    }
    sS[r] = s; sTt[r] = t;
  }
  __syncthreads();

  // ---- row softmax (diag masked) -> g (bf16, K padded with zeros) ----
  for (int r = tid; r < ROWS; r += BLOCK) {
    int bi = r / NF, i = r % NF;
    float s = sS[r];
    const float* tp = sTt + bi * NF;
    float mx = -1e30f;
    for (int j = 0; j < NF; ++j) {
      if (j == i) continue;
      float v = s + tp[j];
      v = (v >= 0.f) ? v : NEG_SLOPE * v;
      mx = fmaxf(mx, v);
    }
    float sum = 0.f;
    for (int j = 0; j < NF; ++j) {
      if (j == i) continue;
      float v = s + tp[j];
      v = (v >= 0.f) ? v : NEG_SLOPE * v;
      sum += __expf(v - mx);
    }
    float inv = 1.f / sum;
    __bf16* gp = sG + r * G_ROW;
    for (int j = 0; j < NF; ++j) {
      if (j == i) { gp[j] = (__bf16)0.f; continue; }
      float v = s + tp[j];
      v = (v >= 0.f) ? v : NEG_SLOPE * v;
      gp[j] = (__bf16)(__expf(v - mx) * inv);
    }
    for (int j = NF; j < DD; ++j) gp[j] = (__bf16)0.f;   // K padding
  }
  __syncthreads();

  for (int layer = 0; layer < 3; ++layer) {
    // ===== Phase 1: h_out^T = W_out[f] @ h^T  ->  hoT[bi][d][f] =====
    for (int task = wv; task < NF * 4; task += NWAVES) {
      int f = task >> 2, mt = task & 3;
      const float* rowp = sH + (bic * NF + f) * SH_ROW;
      v16bf bfr[2];
      #pragma unroll
      for (int ks = 0; ks < 2; ++ks) {
        v16bf v = b_frag_f32(rowp + 32 * ks + 16 * khalf);
        bfr[ks] = biok ? v : vzero;
      }
      v8f acc = {};
      #pragma unroll
      for (int ks = 0; ks < 2; ++ks) {
        const v16bf* ap =
            (const v16bf*)(AswOut + (size_t)((((f * 4 + mt) * 2 + ks) * 32) + lane) * 16);
        acc = wmma_bf16(*ap, bfr[ks], acc);
      }
      if (biok) {
        int d0 = 16 * mt + 8 * khalf;
        __bf16* hb = hoT + bic * HO_SLAB + f;
        #pragma unroll
        for (int i = 0; i < 8; ++i) hb[(d0 + i) * HO_J] = (__bf16)acc[i];
      }
    }
    __syncthreads();

    // ===== Phase 2 (WMMA): aggr = g @ h_out  (K padded to 64) -> sT2 =====
    for (int task = wv; task < TB * 3; task += NWAVES) {
      int bi = task / 3, mt = task % 3;              // mt: f-row tiles (40 -> 48)
      int fr = 16 * mt + lmod;
      int frc = (fr < NF) ? fr : NF - 1;             // clamp (garbage rows unstored)
      const __bf16* growp = sG + (bi * NF + frc) * G_ROW;
      v16bf af[2];
      #pragma unroll
      for (int ks = 0; ks < 2; ++ks) af[ks] = a_frag_bf16(growp, ks, khalf);
      const __bf16* hob = hoT + bi * HO_SLAB;
      #pragma unroll
      for (int nc = 0; nc < 4; ++nc) {
        int d = 16 * nc + lmod;                      // B-lane column
        v8f acc = {};
        #pragma unroll
        for (int ks = 0; ks < 2; ++ks) {
          v16bf bf = frag16_bf16(hob + d * HO_J + 32 * ks + 16 * khalf);
          acc = wmma_bf16(af[ks], bf, acc);
        }
        #pragma unroll
        for (int i = 0; i < 8; ++i) {
          int fo = 16 * mt + i + 8 * khalf;          // D-row = field
          if (fo < NF) sT2[(bi * NF + fo) * T2_ROW + d] = (__bf16)acc[i];
        }
      }
    }
    __syncthreads();

    // ===== Phase 3: a^T = W_in[f] @ aggr^T + bias_p  (in-place in sT2) =====
    for (int f = wv; f < NF; f += NWAVES) {          // wave owns field f entirely
      const __bf16* rowp = sT2 + (bic * NF + f) * T2_ROW;
      v16bf bfr[2];
      #pragma unroll
      for (int ks = 0; ks < 2; ++ks) {
        v16bf v = frag16_bf16(rowp + 32 * ks + 16 * khalf);
        bfr[ks] = biok ? v : vzero;                  // reads complete before stores
      }
      #pragma unroll
      for (int mt = 0; mt < 4; ++mt) {
        v8f acc = {};
        #pragma unroll
        for (int ks = 0; ks < 2; ++ks) {
          const v16bf* ap =
              (const v16bf*)(AswIn + (size_t)((((f * 4 + mt) * 2 + ks) * 32) + lane) * 16);
          acc = wmma_bf16(*ap, bfr[ks], acc);
        }
        if (biok) {
          int d0 = 16 * mt + 8 * khalf;
          v8bf pk;
          #pragma unroll
          for (int i = 0; i < 8; ++i) pk[i] = (__bf16)(acc[i] + bias_p[d0 + i]);
          *(v8bf*)(sT2 + (bic * NF + f) * T2_ROW + d0) = pk;   // contiguous 16B store
        }
      }
    }
    __syncthreads();

    // ===== Phase 4: fused GRU GEMMs + gates + residual (in-place h) =====
    for (int st = wv; st < ROWS / 16; st += NWAVES) {  // wave owns its 16 rows
      int arow = st * 16 + lmod;
      v16bf Aa[2], Ah[2];
      #pragma unroll
      for (int ks = 0; ks < 2; ++ks) {
        Aa[ks] = a_frag_bf16(sT2 + arow * T2_ROW, ks, khalf);  // a (with bias_p)
        Ah[ks] = a_frag_f32(sH + arow * SH_ROW, ks, khalf);    // h
      }
      #pragma unroll
      for (int dc = 0; dc < 4; ++dc) {
        v8f accr = {}, accz = {}, accin = {}, acchn = {};
        #pragma unroll
        for (int ks = 0; ks < 2; ++ks) {
          const __bf16* pih = BswIh + (size_t)(((dc * 2 + ks) * 32) + lane) * 16;
          const __bf16* phh = BswHh + (size_t)(((dc * 2 + ks) * 32) + lane) * 16;
          v16bf bir = *(const v16bf*)(pih);
          v16bf bhr = *(const v16bf*)(phh);
          v16bf biz = *(const v16bf*)(pih + 4 * 1024);   // nc += 4 per gate
          v16bf bhz = *(const v16bf*)(phh + 4 * 1024);
          v16bf bin = *(const v16bf*)(pih + 8 * 1024);
          v16bf bhn = *(const v16bf*)(phh + 8 * 1024);
          accr  = wmma_bf16(Aa[ks], bir, accr);   // i_r
          accr  = wmma_bf16(Ah[ks], bhr, accr);   // + h_r
          accz  = wmma_bf16(Aa[ks], biz, accz);   // i_z
          accz  = wmma_bf16(Ah[ks], bhz, accz);   // + h_z
          accin = wmma_bf16(Aa[ks], bin, accin);  // i_n
          acchn = wmma_bf16(Ah[ks], bhn, acchn);  // h_n
        }
        int d = 16 * dc + lmod;
        float br   = b_ih[d] + b_hh[d];
        float bz   = b_ih[DD + d] + b_hh[DD + d];
        float bi_n = b_ih[2 * DD + d];
        float bh_n = b_hh[2 * DD + d];
        #pragma unroll
        for (int i = 0; i < 8; ++i) {
          int m = st * 16 + i + 8 * khalf;
          float r = 1.f / (1.f + __expf(-(accr[i] + br)));
          float z = 1.f / (1.f + __expf(-(accz[i] + bz)));
          float n = tanhf(accin[i] + bi_n + r * (acchn[i] + bh_n));
          float hold = sH[m * SH_ROW + d];
          float fe = feat[tilebase + (size_t)m * DD + d];   // residual (L2 hit)
          float hnew = (1.f - z) * n + z * hold + fe;
          if (layer == 2) out[tilebase + (size_t)m * DD + d] = hnew;
          else            sH[m * SH_ROW + d] = hnew;
        }
      }
    }
    __syncthreads();
  }
}

extern "C" void kernel_launch(void* const* d_in, const int* in_sizes, int n_in,
                              void* d_out, int out_size, void* d_ws, size_t ws_size,
                              hipStream_t stream) {
  const float* feat     = (const float*)d_in[0];   // [B,F,D]
  const float* w_attn   = (const float*)d_in[1];   // [2D]
  const float* W_in     = (const float*)d_in[2];   // [F,D,D]
  const float* W_out    = (const float*)d_in[3];   // [F,D,D]
  const float* bias_p   = (const float*)d_in[4];   // [D]
  const float* gru_w_ih = (const float*)d_in[5];   // [3D,D]
  const float* gru_w_hh = (const float*)d_in[6];   // [3D,D]
  const float* gru_b_ih = (const float*)d_in[7];   // [3D]
  const float* gru_b_hh = (const float*)d_in[8];   // [3D]
  __bf16* ws = (__bf16*)d_ws;

  int B = in_sizes[0] / (NF * DD);                 // 8192

  fignn_prep_weights<<<(WS_TOTAL + 255) / 256, 256, 0, stream>>>(W_out, W_in, gru_w_ih,
                                                                 gru_w_hh, ws);
  hipFuncSetAttribute((const void*)fignn_fused,
                      hipFuncAttributeMaxDynamicSharedMemorySize, LDS_BYTES);
  fignn_fused<<<B / TB, BLOCK, LDS_BYTES, stream>>>(feat, w_attn, bias_p, gru_b_ih,
                                                    gru_b_hh, ws, (float*)d_out);
}
